// FrameTransformer_47588237640237
// MI455X (gfx1250) — matrix operand order repaired and checked by reading, not compiled, not run on hardware
//
#include <hip/hip_runtime.h>
#include <cstddef>
#include <cstdint>

// ---------------- problem constants ----------------
#define BB   2
#define CC   8
#define FF   1024
#define WW   512
#define HH   8
#define HDIM 128
#define EXPN 4
#define EPSV 1e-8f

typedef __attribute__((ext_vector_type(16))) __bf16 v16bf;
typedef __attribute__((ext_vector_type(8)))  float  v8f;
typedef int v4i __attribute__((vector_size(4 * sizeof(int))));

#define NF ((size_t)BB * CC * FF * WW)   // 8,388,608

static __device__ __forceinline__ v8f wmma_bf16(const v16bf& a, const v16bf& b, const v8f& c) {
  return __builtin_amdgcn_wmma_f32_16x16x32_bf16(false, a, false, b, (short)0, c, false, false);
}

// ---- async global->LDS 16B copy (gfx1250 path), with safe fallback ----
#if defined(__gfx1250__) && __has_builtin(__builtin_amdgcn_global_load_async_to_lds_b128) && \
    __has_builtin(__builtin_amdgcn_s_wait_asynccnt)
#define HAS_ASYNC_LDS 1
#else
#define HAS_ASYNC_LDS 0
#endif

static __device__ __forceinline__ void cp16_g2l(const __bf16* gsrc, __bf16* ldst) {
#if HAS_ASYNC_LDS
  __builtin_amdgcn_global_load_async_to_lds_b128(
      (__attribute__((address_space(1))) v4i*)(void*)(uintptr_t)gsrc,
      (__attribute__((address_space(3))) v4i*)(void*)ldst,
      0, 0);
#else
  *(uint4*)ldst = *(const uint4*)gsrc;
#endif
}
static __device__ __forceinline__ void async_join() {
#if HAS_ASYNC_LDS
  __builtin_amdgcn_s_wait_asynccnt(0);
#endif
}

// Fragment layout inversion (16-bit operands, one 32-K chunk):
//  A-operand: lane = (m&15) + 16*((k>>3)&1), elem = 2*((k>>4)*4 + ((k&7)>>1)) + (k&1)
//             -> 8 consecutive k (8-aligned) = 16 contiguous bytes of one lane
//  B-operand: lane = (n&15) + 16*((k>>3)&1), elem = 8*(k>>4) + (k&7) (same k order as A)

// =====================================================================
// f32 -> bf16 tensor convert (weights / activations), 4 elems per thread
// =====================================================================
__global__ void cvt_f32_to_bf16(const float* __restrict__ in, __bf16* __restrict__ out,
                                size_t n) {
  size_t idx = ((size_t)blockIdx.x * blockDim.x + threadIdx.x) * 4;
  if (idx >= n) return;
  float4 v = *(const float4*)&in[idx];
  __bf16* o = &out[idx];
  o[0] = (__bf16)v.x; o[1] = (__bf16)v.y; o[2] = (__bf16)v.z; o[3] = (__bf16)v.w;
}

// =====================================================================
// MultichannelLayerNorm over feature axis
// grid: (WW/64, BB*CC), block 256
// =====================================================================
__global__ void mcln_kernel(const float* __restrict__ x,
                            const float* __restrict__ g,
                            const float* __restrict__ bta,
                            float* __restrict__ out) {
  const int bc = blockIdx.y;
  const int c  = bc % CC;
  const int w0 = blockIdx.x * 64;
  const int wx = threadIdx.x % 64;
  const int fy = threadIdx.x / 64;
  const float* xp = x + (size_t)bc * FF * WW + w0;

  float s = 0.f, s2 = 0.f;
  for (int f = fy; f < FF; f += 4) {
    float v = xp[(size_t)f * WW + wx];
    s += v; s2 += v * v;
  }
  __shared__ float rs[4][64], rs2[4][64], mu[64], rv[64];
  rs[fy][wx] = s; rs2[fy][wx] = s2;
  __syncthreads();
  if (fy == 0) {
    float S  = rs[0][wx] + rs[1][wx] + rs[2][wx] + rs[3][wx];
    float S2 = rs2[0][wx] + rs2[1][wx] + rs2[2][wx] + rs2[3][wx];
    float m  = S / (float)FF;
    float va = S2 / (float)FF - m * m;
    mu[wx] = m;
    rv[wx] = rsqrtf(va + EPSV);
  }
  __syncthreads();
  float* op = out + (size_t)bc * FF * WW + w0;
  const float* gp = g   + (size_t)c * FF;
  const float* bp = bta + (size_t)c * FF;
  for (int f = fy; f < FF; f += 4) {
    float v = xp[(size_t)f * WW + wx];
    op[(size_t)f * WW + wx] = (v - mu[wx]) * rv[wx] * gp[f] + bp[f];
  }
}

// =====================================================================
// GEMM: Y[bc][m][n] = sum_k Wb[c][m][k] * X[bc][k][n]
// Wb: pre-converted bf16 -> A tiles staged as raw 16B (async) copies.
// X:  f32 activations, converted during staging.
// Tile 64x64, BK=32; 256 threads = 8 waves; grid (N/64, M/64, BB*CC)
// =====================================================================
__global__ void gemm_wmma_bf16(const __bf16* __restrict__ Wb,  // [CC, M, K] bf16
                               const float*  __restrict__ X,   // [BB*CC, K, N] f32
                               float*        __restrict__ Y,   // [BB*CC, M, N]
                               int M, int K, int N) {
  const int bc = blockIdx.z;
  const int c  = bc % CC;
  const int m0 = blockIdx.y * 64;
  const int n0 = blockIdx.x * 64;
  const __bf16* A  = Wb + (size_t)c * M * K;
  const float*  Bm = X  + (size_t)bc * K * N;
  float*        Yp = Y  + (size_t)bc * M * N;

  __shared__ __align__(32) __bf16 ASw[4][32][16];
  __shared__ __align__(32) __bf16 BSw[4][32][16];

  const int tid  = threadIdx.x;
  const int lane = tid & 31;
  const int wv   = tid >> 5;
  const int wm   = wv & 3;
  const int wn   = wv >> 2;
  const int hl   = lane >> 4;
  const int l16  = lane & 15;

  v8f acc0 = {}; v8f acc1 = {};

  for (int k0 = 0; k0 < K; k0 += 32) {
    // ---- stage A: 64 rows x 4 chunks of 8 k = 256 16B copies, 1/thread ----
    {
      int m = tid >> 2, t = tid & 3;          // k = 8t..8t+7
      cp16_g2l(&A[(size_t)(m0 + m) * K + k0 + 8 * t],
               &ASw[m >> 4][(m & 15) + 16 * (t & 1)][8 * (t >> 1)]);
    }
    // ---- stage B: f32, 4 strided k per item; batch loads, then cvt+store ----
    float bv[2][4];
#pragma unroll
    for (int it = 0; it < 2; ++it) {
      int i  = tid + it * 256;
      int k4 = (i >> 6) << 2;
      int n  = i & 63;
      const float* bp = &Bm[(size_t)(k0 + k4) * N + n0 + n];
      bv[it][0] = bp[0];
      bv[it][1] = bp[(size_t)N];
      bv[it][2] = bp[(size_t)2 * N];
      bv[it][3] = bp[(size_t)3 * N];
    }
#pragma unroll
    for (int it = 0; it < 2; ++it) {
      int i  = tid + it * 256;
      int k4 = (i >> 6) << 2;
      int n  = i & 63;
      __bf16* dst = &BSw[n >> 4][(n & 15) + 16 * ((k4 >> 3) & 1)][8 * (k4 >> 4) + (k4 & 7)];
      dst[0] = (__bf16)bv[it][0]; dst[1] = (__bf16)bv[it][1];
      dst[2] = (__bf16)bv[it][2]; dst[3] = (__bf16)bv[it][3];
    }
    async_join();
    __syncthreads();

    if (k0 + 32 < K) {
      __builtin_prefetch(&A[(size_t)(m0 + (tid >> 2)) * K + k0 + 32], 0, 1);
      __builtin_prefetch(&Bm[(size_t)(k0 + 32 + (tid >> 4)) * N + n0 + ((tid & 15) << 2)], 0, 1);
    }

    v16bf fa  = *(const v16bf*)ASw[wm][lane];
    v16bf fb0 = *(const v16bf*)BSw[wn * 2][lane];
    v16bf fb1 = *(const v16bf*)BSw[wn * 2 + 1][lane];
    acc0 = wmma_bf16(fa, fb0, acc0);
    acc1 = wmma_bf16(fa, fb1, acc1);
    __syncthreads();
  }

  for (int r = 0; r < 8; ++r) {
    int row  = m0 + wm * 16 + hl * 8 + r;
    int col0 = n0 + wn * 32 + l16;
    Yp[(size_t)row * N + col0]      = acc0[r];
    Yp[(size_t)row * N + col0 + 16] = acc1[r];
  }
}

// =====================================================================
// Flash attention, bf16 Q/K/V in [BB*CC*HH, WW, HDIM].
// Q/K tiles: raw 16B (async) copies into fragment layout.
// grid: (WW/64, BB*CC*HH), block 256
// =====================================================================
__global__ void flash_attn(const __bf16* __restrict__ Q,
                           const __bf16* __restrict__ Kt,
                           const __bf16* __restrict__ V,
                           float* __restrict__ O) {
  const int bh = blockIdx.y;
  const int q0 = blockIdx.x * 64;
  const __bf16* Qp = Q  + (size_t)bh * WW * HDIM;
  const __bf16* Kp = Kt + (size_t)bh * WW * HDIM;
  const __bf16* Vp = V  + (size_t)bh * WW * HDIM;
  float*        Op = O  + (size_t)bh * WW * HDIM;

  __shared__ __align__(32) __bf16 QSw[4][4][32][16];
  __shared__ __align__(32) __bf16 KSw[2][4][32][16];
  __shared__ __align__(32) __bf16 VSw[8][32][16];
  __shared__ __align__(32) __bf16 PSw[4][32][16];
  __shared__ float Ss[64][36];
  __shared__ float mrow[64], lrow[64], arow[64];

  const int tid  = threadIdx.x;
  const int lane = tid & 31;
  const int wv   = tid >> 5;
  const int wm   = wv & 3;
  const int wn   = wv >> 2;
  const int hl   = lane >> 4;
  const int l16  = lane & 15;
  const float scale = 0.03125f;  // 1/sqrt(F)

  // ---- stage Q (64x128, A-fragment layout): 1024 16B copies ----
#pragma unroll
  for (int it = 0; it < 4; ++it) {
    int i = tid + it * 256;
    int m = i >> 4, t = i & 15;            // d = 8t..8t+7
    cp16_g2l(&Qp[(size_t)(q0 + m) * HDIM + 8 * t],
             &QSw[m >> 4][t >> 2][(m & 15) + 16 * (t & 1)][8 * ((t & 3) >> 1)]);
  }
  if (tid < 64) { mrow[tid] = -3.0e38f; lrow[tid] = 0.f; }
  v8f oacc[4] = {};
  async_join();
  __syncthreads();

  for (int u0 = 0; u0 < WW; u0 += 32) {
    // ---- stage K tile (32x128, B-fragment layout): 512 16B copies ----
#pragma unroll
    for (int it = 0; it < 2; ++it) {
      int i   = tid + it * 256;
      int key = i >> 4, t = i & 15;        // d = 8t..8t+7
      cp16_g2l(&Kp[(size_t)(u0 + key) * HDIM + 8 * t],
               &KSw[key >> 4][t >> 2][(key & 15) + 16 * (t & 1)][8 * ((t >> 1) & 1)]);
    }
    // ---- stage V tile: B-fragments over K=u; 4 strided bf16 per 8B store --
#pragma unroll
    for (int it = 0; it < 4; ++it) {
      int i  = tid + it * 256;
      int u4 = (i >> 7) << 2;              // u = u4..u4+3
      int d  = i & 127;
      const __bf16* vp = &Vp[(size_t)(u0 + u4) * HDIM + d];
      __bf16 b0 = vp[0];
      __bf16 b1 = vp[HDIM];
      __bf16 b2 = vp[2 * HDIM];
      __bf16 b3 = vp[3 * HDIM];
      __bf16* dst = &VSw[d >> 4][(d & 15) + 16 * ((u4 >> 3) & 1)][8 * (u4 >> 4) + (u4 & 7)];
      dst[0] = b0; dst[1] = b1; dst[2] = b2; dst[3] = b3;
    }
    async_join();
    __syncthreads();

    // ---- S = Q * K^T ----
    v8f s = {};
#pragma unroll
    for (int kc = 0; kc < 4; ++kc) {
      v16bf aq = *(const v16bf*)QSw[wm][kc][lane];
      v16bf bk = *(const v16bf*)KSw[wn][kc][lane];
      s = wmma_bf16(aq, bk, s);
    }
#pragma unroll
    for (int r = 0; r < 8; ++r)
      Ss[wm * 16 + hl * 8 + r][wn * 16 + l16] = s[r] * scale;
    __syncthreads();

    // ---- online softmax; P written directly in A-fragment layout ----
    if (tid < 64) {
      float mo = mrow[tid], mx = mo;
#pragma unroll
      for (int u = 0; u < 32; ++u) mx = fmaxf(mx, Ss[tid][u]);
      float al = __expf(mo - mx);
      float sum = 0.f;
#pragma unroll
      for (int u4 = 0; u4 < 32; u4 += 4) {
        float p0v = __expf(Ss[tid][u4 + 0] - mx);
        float p1v = __expf(Ss[tid][u4 + 1] - mx);
        float p2v = __expf(Ss[tid][u4 + 2] - mx);
        float p3v = __expf(Ss[tid][u4 + 3] - mx);
        sum += p0v + p1v + p2v + p3v;
        int ah = (u4 >> 3) & 1;
        int pp = (u4 >> 4) * 4 + ((u4 & 7) >> 1);
        __bf16* dst = &PSw[tid >> 4][(tid & 15) + 16 * ah][2 * pp];
        dst[0] = (__bf16)p0v; dst[1] = (__bf16)p1v;
        dst[2] = (__bf16)p2v; dst[3] = (__bf16)p3v;
      }
      mrow[tid] = mx;
      lrow[tid] = lrow[tid] * al + sum;
      arow[tid] = al;
    }
    __syncthreads();

    // ---- rescale O, then O += P * V ----
#pragma unroll
    for (int j = 0; j < 4; ++j)
#pragma unroll
      for (int r = 0; r < 8; ++r)
        oacc[j][r] *= arow[wm * 16 + hl * 8 + r];

    v16bf ap = *(const v16bf*)PSw[wm][lane];
#pragma unroll
    for (int j = 0; j < 4; ++j) {
      v16bf bv = *(const v16bf*)VSw[wn * 4 + j][lane];
      oacc[j] = wmma_bf16(ap, bv, oacc[j]);
    }
    __syncthreads();
  }

#pragma unroll
  for (int j = 0; j < 4; ++j)
#pragma unroll
    for (int r = 0; r < 8; ++r) {
      int row = wm * 16 + hl * 8 + r;
      int col = wn * 64 + j * 16 + l16;
      Op[(size_t)(q0 + row) * HDIM + col] = oacc[j][r] / lrow[row];
    }
}

// =====================================================================
// elementwise / small kernels
// =====================================================================
__global__ void glu_combine(float* __restrict__ h, const float* __restrict__ zz) {
  size_t idx = (size_t)blockIdx.x * blockDim.x + threadIdx.x;
  if (idx >= NF) return;
  int w = (int)(idx % WW); size_t r = idx / WW;
  int f = (int)(r % FF);  size_t bc = r / FF;
  const float* zp = zz + bc * (size_t)(2 * FF * WW);
  float a = zp[(size_t)f * WW + w];
  float g = zp[(size_t)(FF + f) * WW + w];
  h[idx] += a * (1.f / (1.f + __expf(-g)));
}

__global__ void dwconv_kernel(const float* __restrict__ x, const float* __restrict__ ker,
                              float* __restrict__ out, int ksz, int pad) {
  size_t idx = (size_t)blockIdx.x * blockDim.x + threadIdx.x;
  if (idx >= NF) return;
  int w = (int)(idx % WW); size_t r = idx / WW;
  size_t r2 = r / FF;
  int c = (int)(r2 % CC);
  const float* xp = x + (idx - w);
  const float* kp = ker + c * ksz;
  float acc = 0.f;
  for (int t = 0; t < ksz; ++t) {
    int ww = w + t - pad;
    if (ww >= 0 && ww < WW) acc += xp[ww] * kp[t];
  }
  out[idx] = acc;
}

__global__ void chconv3(const float* __restrict__ t, const float* __restrict__ cw,
                        const float* __restrict__ cb, float* __restrict__ out) {
  size_t idx = (size_t)blockIdx.x * blockDim.x + threadIdx.x;
  if (idx >= NF) return;
  int w = (int)(idx % WW); size_t r = idx / WW;
  int f = (int)(r % FF);  size_t r2 = r / FF;
  int o = (int)(r2 % CC); int b = (int)(r2 / CC);
  float acc = cb[o];
#pragma unroll
  for (int i = 0; i < CC; ++i) {
    const float* tp = t + ((size_t)(b * CC + i) * FF + f) * WW;
    const float* wp = cw + (o * CC + i) * 3;
    float v0 = (w > 0)      ? tp[w - 1] : 0.f;
    float v1 = tp[w];
    float v2 = (w < WW - 1) ? tp[w + 1] : 0.f;
    acc += v0 * wp[0] + v1 * wp[1] + v2 * wp[2];
  }
  out[idx] = acc;
}

__global__ void add_sqrelu2(float* __restrict__ h, const float* __restrict__ t1,
                            const float* __restrict__ t2) {
  size_t idx = (size_t)blockIdx.x * blockDim.x + threadIdx.x;
  if (idx >= NF) return;
  float a = fmaxf(t1[idx], 0.f);
  float b = fmaxf(t2[idx], 0.f);
  h[idx] += a * a + b * b;
}

__global__ void add_plain(float* __restrict__ h, const float* __restrict__ t) {
  size_t idx = (size_t)blockIdx.x * blockDim.x + threadIdx.x;
  if (idx >= NF) return;
  h[idx] += t[idx];
}

// [BB*CC, F, W] f32 -> [BB*CC*HH, W, HDIM] bf16, optional RoPE
__global__ void reorder_rope(const float* __restrict__ t, __bf16* __restrict__ out,
                             int do_rope) {
  size_t idx = (size_t)blockIdx.x * blockDim.x + threadIdx.x;
  size_t total = (size_t)BB * CC * HH * WW * (HDIM / 2);
  if (idx >= total) return;
  int j = (int)(idx % (HDIM / 2)); size_t r = idx / (HDIM / 2);
  int w = (int)(r % WW); size_t bch = r / WW;
  int hh = (int)(bch % HH); size_t bc = bch / HH;
  const float* tp = t + bc * (size_t)(FF * WW);
  float x1 = tp[(size_t)(hh * HDIM + 2 * j) * WW + w];
  float x2 = tp[(size_t)(hh * HDIM + 2 * j + 1) * WW + w];
  float o1 = x1, o2 = x2;
  if (do_rope) {
    float inv = __powf(10000.f, -(float)(2 * j) / (float)HDIM);
    float fr  = (float)w * inv;
    float cs = __cosf(fr), sn = __sinf(fr);
    o1 = x1 * cs - x2 * sn;
    o2 = x2 * cs + x1 * sn;
  }
  __bf16* op = out + bch * (size_t)(WW * HDIM) + (size_t)w * HDIM;
  op[2 * j]     = (__bf16)o1;
  op[2 * j + 1] = (__bf16)o2;
}

// [BB*CC*HH, W, HDIM] f32 -> [BB*CC, F, W] f32
__global__ void reorder_back(const float* __restrict__ a, float* __restrict__ out) {
  size_t idx = (size_t)blockIdx.x * blockDim.x + threadIdx.x;
  if (idx >= NF) return;
  int w = (int)(idx % WW); size_t r = idx / WW;
  int f = (int)(r % FF);  size_t bc = r / FF;
  int hh = f / HDIM, d = f % HDIM;
  out[idx] = a[((size_t)(bc * HH + hh)) * (WW * HDIM) + (size_t)w * HDIM + d];
}

__global__ void ffn1(const float* __restrict__ z, const float* __restrict__ c3,
                     float* __restrict__ e) {
  size_t idx = (size_t)blockIdx.x * blockDim.x + threadIdx.x;
  size_t total = (size_t)BB * CC * EXPN * WW;
  if (idx >= total) return;
  int w = (int)(idx % WW); size_t r = idx / WW;
  int xch = (int)(r % EXPN); size_t bc = r / EXPN;
  int c = (int)(bc % CC);
  const float* zp = z + bc * (size_t)(FF * WW);
  const float* wp = c3 + ((size_t)c * EXPN + xch) * FF;
  float acc = 0.f;
  for (int f = 0; f < FF; ++f) acc += wp[f] * zp[(size_t)f * WW + w];
  float rr = fmaxf(acc, 0.f);
  e[idx] = rr * rr;
}

__global__ void ffn2(float* __restrict__ h, const float* __restrict__ e,
                     const float* __restrict__ c4) {
  size_t idx = (size_t)blockIdx.x * blockDim.x + threadIdx.x;
  if (idx >= NF) return;
  int w = (int)(idx % WW); size_t r = idx / WW;
  int f = (int)(r % FF);  size_t bc = r / FF;
  int c = (int)(bc % CC);
  const float* ep = e + bc * (size_t)(EXPN * WW);
  const float* wp = c4 + ((size_t)c * FF + f) * EXPN;
  float acc = 0.f;
#pragma unroll
  for (int x = 0; x < EXPN; ++x) acc += wp[x] * ep[(size_t)x * WW + w];
  h[idx] += acc;
}

__global__ void write_out(const float* __restrict__ x, const float* __restrict__ h,
                          float* __restrict__ out) {
  size_t idx = (size_t)blockIdx.x * blockDim.x + threadIdx.x;
  size_t total1 = (size_t)BB * 2 * CC * FF * WW;
  size_t total  = total1 + NF;
  if (idx >= total) return;
  if (idx < total1) {
    int w = (int)(idx % WW); size_t r = idx / WW;
    int f = (int)(r % FF);  size_t r2 = r / FF;
    int cc = (int)(r2 % (2 * CC)); int b = (int)(r2 / (2 * CC));
    const float* src = (cc < CC)
        ? x + ((size_t)(b * CC + cc) * FF + f) * WW + w
        : h + ((size_t)(b * CC + cc - CC) * FF + f) * WW + w;
    out[idx] = *src;
  } else {
    out[idx] = h[idx - total1];
  }
}

// =====================================================================
extern "C" void kernel_launch(void* const* d_in, const int* in_sizes, int n_in,
                              void* d_out, int out_size, void* d_ws, size_t ws_size,
                              hipStream_t stream) {
  const float* x     = (const float*)d_in[0];
  const float* ln1g  = (const float*)d_in[1];  const float* ln1b = (const float*)d_in[2];
  const float* ln2g  = (const float*)d_in[3];  const float* ln2b = (const float*)d_in[4];
  const float* ln3g  = (const float*)d_in[5];  const float* ln3b = (const float*)d_in[6];
  const float* ln4g  = (const float*)d_in[7];  const float* ln4b = (const float*)d_in[8];
  const float* ln5g  = (const float*)d_in[9];  const float* ln5b = (const float*)d_in[10];
  const float* glu_w = (const float*)d_in[11];
  const float* c1a_w = (const float*)d_in[12];
  const float* c1b_w = (const float*)d_in[13];
  const float* c2_w  = (const float*)d_in[14];
  const float* aq_w  = (const float*)d_in[15]; const float* aq_cw = (const float*)d_in[16]; const float* aq_cb = (const float*)d_in[17];
  const float* ak_w  = (const float*)d_in[18]; const float* ak_cw = (const float*)d_in[19]; const float* ak_cb = (const float*)d_in[20];
  const float* av_w  = (const float*)d_in[21]; const float* av_cw = (const float*)d_in[22]; const float* av_cb = (const float*)d_in[23];
  const float* ao_w  = (const float*)d_in[24];
  const float* c3_w  = (const float*)d_in[25];
  const float* c4_w  = (const float*)d_in[26];
  float* out = (float*)d_out;

  float* ws  = (float*)d_ws;
  float* h   = ws;                  // NF f32
  float* z   = ws + 1 * NF;         // NF f32 (also reused as attn output)
  float* s0  = ws + 2 * NF;         // 2*NF f32
  float* s0a = s0;
  float* s0b = s0 + NF;
  float* e   = ws + 4 * NF;         // 32768 f32
  __bf16* bb = (__bf16*)(ws + 4 * NF + 32768);
  __bf16* qr   = bb;                // NF bf16
  __bf16* kr   = bb + 1 * NF;
  __bf16* vr   = bb + 2 * NF;
  __bf16* wbglu = bb + 3 * NF;      // 2*NF bf16
  __bf16* wbc1a = bb + 5 * NF;      // NF bf16 each
  __bf16* wbq   = bb + 6 * NF;
  __bf16* wbk   = bb + 7 * NF;
  __bf16* wbv   = bb + 8 * NF;
  __bf16* wbo   = bb + 9 * NF;

  const dim3 blk(256);
  const dim3 gEw((unsigned)((NF + 255) / 256));
  const dim3 gLn(WW / 64, BB * CC);
  const dim3 gG2048(WW / 64, 2 * FF / 64, BB * CC);
  const dim3 gG1024(WW / 64, FF / 64, BB * CC);
  const dim3 gRope((unsigned)(((size_t)BB * CC * HH * WW * (HDIM / 2) + 255) / 256));
  const dim3 gAttn(WW / 64, BB * CC * HH);
  const dim3 gF1((unsigned)(((size_t)BB * CC * EXPN * WW + 255) / 256));
  const dim3 gOut((unsigned)(((size_t)BB * 3 * CC * FF * WW + 255) / 256));
  const dim3 gCvt1((unsigned)((NF / 4 + 255) / 256));
  const dim3 gCvt2((unsigned)((2 * NF / 4 + 255) / 256));

  // ---- one-time (per launch) weight conversions to bf16 ----
  cvt_f32_to_bf16<<<gCvt2, blk, 0, stream>>>(glu_w, wbglu, 2 * NF);
  cvt_f32_to_bf16<<<gCvt1, blk, 0, stream>>>(c1a_w, wbc1a, NF);
  cvt_f32_to_bf16<<<gCvt1, blk, 0, stream>>>(aq_w,  wbq,   NF);
  cvt_f32_to_bf16<<<gCvt1, blk, 0, stream>>>(ak_w,  wbk,   NF);
  cvt_f32_to_bf16<<<gCvt1, blk, 0, stream>>>(av_w,  wbv,   NF);
  cvt_f32_to_bf16<<<gCvt1, blk, 0, stream>>>(ao_w,  wbo,   NF);

  (void)hipMemcpyAsync(h, x, NF * sizeof(float), hipMemcpyDeviceToDevice, stream);

  // ---- GLU block ----
  mcln_kernel<<<gLn, blk, 0, stream>>>(h, ln1g, ln1b, z);
  gemm_wmma_bf16<<<gG2048, blk, 0, stream>>>(wbglu, z, s0, 2 * FF, FF, WW);
  glu_combine<<<gEw, blk, 0, stream>>>(h, s0);

  // ---- parallel linear + depthwise conv, squared-relu ----
  mcln_kernel<<<gLn, blk, 0, stream>>>(h, ln2g, ln2b, z);
  gemm_wmma_bf16<<<gG1024, blk, 0, stream>>>(wbc1a, z, s0a, FF, FF, WW);
  dwconv_kernel<<<gEw, blk, 0, stream>>>(z, c1b_w, s0b, 3, 1);
  add_sqrelu2<<<gEw, blk, 0, stream>>>(h, s0a, s0b);

  // ---- depthwise frame conv k=9 ----
  mcln_kernel<<<gLn, blk, 0, stream>>>(h, ln3g, ln3b, z);
  dwconv_kernel<<<gEw, blk, 0, stream>>>(z, c2_w, s0a, 9, 4);
  add_plain<<<gEw, blk, 0, stream>>>(h, s0a);

  // ---- attention ----
  mcln_kernel<<<gLn, blk, 0, stream>>>(h, ln4g, ln4b, z);
  gemm_wmma_bf16<<<gG1024, blk, 0, stream>>>(wbq, z, s0a, FF, FF, WW);
  chconv3<<<gEw, blk, 0, stream>>>(s0a, aq_cw, aq_cb, s0b);
  reorder_rope<<<gRope, blk, 0, stream>>>(s0b, qr, 1);
  gemm_wmma_bf16<<<gG1024, blk, 0, stream>>>(wbk, z, s0a, FF, FF, WW);
  chconv3<<<gEw, blk, 0, stream>>>(s0a, ak_cw, ak_cb, s0b);
  reorder_rope<<<gRope, blk, 0, stream>>>(s0b, kr, 1);
  gemm_wmma_bf16<<<gG1024, blk, 0, stream>>>(wbv, z, s0a, FF, FF, WW);
  chconv3<<<gEw, blk, 0, stream>>>(s0a, av_cw, av_cb, s0b);
  reorder_rope<<<gRope, blk, 0, stream>>>(s0b, vr, 0);
  flash_attn<<<gAttn, blk, 0, stream>>>(qr, kr, vr, z);   // z = attn out
  reorder_back<<<gEw, blk, 0, stream>>>(z, s0a);
  gemm_wmma_bf16<<<gG1024, blk, 0, stream>>>(wbo, s0a, s0b, FF, FF, WW);
  add_plain<<<gEw, blk, 0, stream>>>(h, s0b);

  // ---- bottleneck FFN ----
  mcln_kernel<<<gLn, blk, 0, stream>>>(h, ln5g, ln5b, z);
  ffn1<<<gF1, blk, 0, stream>>>(z, c3_w, e);
  ffn2<<<gEw, blk, 0, stream>>>(h, e, c4_w);

  // ---- outputs ----
  write_out<<<gOut, blk, 0, stream>>>(x, h, out);

  (void)in_sizes; (void)n_in; (void)out_size; (void)ws_size;
}
